// MRNN_13864154432059
// MI455X (gfx1250) — matrix operand order repaired and checked by reading, not compiled, not run on hardware
//
#include <hip/hip_runtime.h>
#include <hip/hip_bf16.h>
#include <stdint.h>

// ---------------------------------------------------------------------------
// 2-layer LSTM for MI455X (gfx1250), bf16 WMMA path.
//   B=64, T=256, IN=H=1024, 4H=4096, K = IN+H = 2048 (x|h concat)
// Per timestep one fused launch (gridDim.y = #active layers) computes, per
// layer, [64 x 4096] = [64 x 2048]*[2048 x 4096] via v_wmma_f32_16x16x32_bf16
// and the fused sigmoid/tanh gate math. A-panel staged through LDS in
// 256-wide K slabs (async-to-LDS when available); bf16 weights L2-resident.
// Workspace use: ~135 MB of d_ws.
// ---------------------------------------------------------------------------

#define B_    64
#define T_    256
#define IN_   1024
#define H_    1024
#define KTOT  2048
#define FOURH 4096
#define KSLAB 256
#define AROWDW 132   // LDS row stride in dwords for 256 bf16 (+4 dw pad)

typedef __attribute__((ext_vector_type(16))) __bf16 v16bf;
typedef __attribute__((ext_vector_type(8)))  float  v8f;
typedef __attribute__((ext_vector_type(4)))  int    v4i;

union ABVec { uint4 q[2]; v16bf v; };

#if __has_builtin(__builtin_amdgcn_global_load_async_to_lds_b128) && \
    __has_builtin(__builtin_amdgcn_s_wait_asynccnt)
#define ASYNC_LDS 1
#else
#define ASYNC_LDS 0
#endif

static __device__ __forceinline__ unsigned short f2bf(float f) {
  union { float f; unsigned int u; } c; c.f = f;
  unsigned int u = c.u;
  unsigned int r = u + 0x7FFFu + ((u >> 16) & 1u);   // round-to-nearest-even
  return (unsigned short)(r >> 16);
}

static __device__ __forceinline__ float sigm_fast(float x) {
  return 1.0f / (1.0f + __expf(-x));
}
static __device__ __forceinline__ float tanh_fast(float x) {
  x = fminf(15.0f, fmaxf(-15.0f, x));
  float e = __expf(2.0f * x);
  return (e - 1.0f) / (e + 1.0f);
}

// --------------------------- prep kernels ----------------------------------

// Build Wcat bf16 [4096][2048] row-major: k<1024 from Wx, k>=1024 from Wh.
// Also bias = bx + bh.
__global__ void prep_weights_kernel(const float* __restrict__ Wx,
                                    const float* __restrict__ Wh,
                                    const float* __restrict__ bx,
                                    const float* __restrict__ bh,
                                    unsigned short* __restrict__ Wcat,
                                    float* __restrict__ bias) {
  long idx = blockIdx.x * (long)blockDim.x + threadIdx.x;
  long total = (long)FOURH * KTOT;
  if (idx < total) {
    int n = (int)(idx >> 11);     // /2048
    int k = (int)(idx & 2047);
    float v = (k < IN_) ? Wx[(long)n * IN_ + k] : Wh[(long)n * H_ + (k - IN_)];
    Wcat[idx] = f2bf(v);
  }
  if (idx < FOURH) bias[idx] = bx[idx] + bh[idx];
}

__global__ void prep_x_kernel(const float* __restrict__ x,
                              unsigned short* __restrict__ xbf, long n) {
  long idx = blockIdx.x * (long)blockDim.x + threadIdx.x;
  if (idx < n) xbf[idx] = f2bf(x[idx]);
}

// slot 0 of each h-sequence <- h0 (bf16); c state <- c0 (fp32)
__global__ void prep_state_kernel(const float* __restrict__ h0,
                                  const float* __restrict__ c0,
                                  unsigned short* __restrict__ h0seq0,
                                  unsigned short* __restrict__ h1seq0,
                                  float* __restrict__ cst) {
  int idx = blockIdx.x * blockDim.x + threadIdx.x;
  if (idx >= B_ * H_) return;
  h0seq0[idx] = f2bf(h0[idx]);
  h1seq0[idx] = f2bf(h0[B_ * H_ + idx]);
  cst[idx]            = c0[idx];
  cst[B_ * H_ + idx]  = c0[B_ * H_ + idx];
}

// --------------------------- fused LSTM step -------------------------------

struct StepJob {
  const unsigned short* Wcat;   // [4096][2048] bf16
  const float* bias;            // [4096] (bx+bh)
  const unsigned short* xbase;  // row b at xbase + b*xstride
  long xstride;
  const unsigned short* hprev;  // [B][H] bf16
  unsigned short* hout;         // [B][H] bf16
  float* c;                     // [B][H] f32 (in/out)
  float* out32;                 // nullable; row b at out32 + b*outStride
  long outStride;
};

// Block = 256 threads = 8 waves; block owns 16 hidden units x 4 gates x 64
// batch rows (64x64 f32 tile). wave w: m-tile = w&3, gates {2*(w>>2)+0,1}.
// blockIdx.y selects the job (layer) for software-pipelined layer overlap.
__global__ __launch_bounds__(256)
void lstm_step_kernel(StepJob jA, StepJob jB) {
  __shared__ unsigned int ldsA[64 * AROWDW];  // 64 rows x 256 bf16 (padded)
  __shared__ float        stage[64 * 65];     // 64 x 64 preact, padded

  const StepJob jb = (blockIdx.y == 0) ? jA : jB;

  const int tid  = threadIdx.x;
  const int lane = tid & 31;
  const int wave = tid >> 5;
  const int j0   = blockIdx.x * 16;

  const int mt = wave & 3;
  const int g0 = (wave >> 2) * 2;

  v8f acc0 = {};
  v8f acc1 = {};

  // A-matrix per-lane addressing (ISA 16-bit A 16x32 layout):
  // lanes 0-15: row M=l, K {0..7,16..23}; lanes 16-31: row M=l-16, K {8..15,24..31}
  const int khalf = lane >> 4;
  const unsigned int* aRow = ldsA + (mt * 16 + (lane & 15)) * AROWDW;

  // B-matrix per-lane addressing (same K order as A):
  // lanes 0-15: col N=l, K {0..7,16..23}; lanes 16-31: col N=l-16, K {8..15,24..31}
  const unsigned short* bRow0 =
      jb.Wcat + (long)(g0 * H_ + j0 + (lane & 15)) * KTOT + (lane >> 4) * 8;
  const unsigned short* bRow1 = bRow0 + (long)H_ * KTOT;

  // cooperative slab load: chunk c = tid + 256*i -> row = wave + 8*i, kelem = lane*8
  for (int slab = 0; slab < KTOT; slab += KSLAB) {
    const unsigned short* srcBase;
    long rstride;
    if (slab < IN_) { srcBase = jb.xbase + slab;        rstride = jb.xstride; }
    else            { srcBase = jb.hprev + (slab - IN_); rstride = H_; }
    srcBase += lane * 8;

    __syncthreads();   // previous slab's ds reads done
#pragma unroll
    for (int i = 0; i < 8; ++i) {
      const int row = wave + 8 * i;
      const uint4* g = (const uint4*)(srcBase + (long)row * rstride);
      unsigned int* l = &ldsA[row * AROWDW + lane * 4];
#if ASYNC_LDS
      __builtin_amdgcn_global_load_async_to_lds_b128(
          (v4i*)(void*)g, (v4i*)(void*)l, 0, 0);
#else
      *(uint4*)l = *g;
#endif
    }
#if ASYNC_LDS
    __builtin_amdgcn_s_wait_asynccnt(0);
#endif
    __syncthreads();   // slab visible to all waves

    // keep next slab's weight rows hot in L2/WGP$
    if (slab + KSLAB < KTOT) {
      __builtin_prefetch(bRow0 + slab + KSLAB, 0, 2);
      __builtin_prefetch(bRow1 + slab + KSLAB, 0, 2);
    }

#pragma unroll
    for (int kk = 0; kk < KSLAB; kk += 32) {
      ABVec a, b0, b1;
      const unsigned int* ap = aRow + (kk >> 1);
      a.q[0] = *(const uint4*)(ap + khalf * 4);
      a.q[1] = *(const uint4*)(ap + 8 + khalf * 4);
      const unsigned short* p0 = bRow0 + slab + kk;
      const unsigned short* p1 = bRow1 + slab + kk;
      b0.q[0] = *(const uint4*)p0;
      b0.q[1] = *(const uint4*)(p0 + 16);
      b1.q[0] = *(const uint4*)p1;
      b1.q[1] = *(const uint4*)(p1 + 16);
      acc0 = __builtin_amdgcn_wmma_f32_16x16x32_bf16(false, a.v, false, b0.v,
                                                     (short)0, acc0, false, false);
      acc1 = __builtin_amdgcn_wmma_f32_16x16x32_bf16(false, a.v, false, b1.v,
                                                     (short)0, acc1, false, false);
    }
  }

  // Stage preactivations: C layout VGPR r: lanes 0-15 -> M=r, lanes 16-31 -> M=8+r.
  // Waves write disjoint stage regions.
  {
    const int rbase = mt * 16 + ((lane >> 4) ? 8 : 0);
    const int colA  = g0 * 16 + (lane & 15);
    const int colB  = (g0 + 1) * 16 + (lane & 15);
#pragma unroll
    for (int r = 0; r < 8; ++r) {
      stage[(rbase + r) * 65 + colA] = acc0[r];
      stage[(rbase + r) * 65 + colB] = acc1[r];
    }
  }
  __syncthreads();

  // Fused gate math: 64 rows x 16 units = 1024 items, 4 per thread.
#pragma unroll
  for (int it = 0; it < 4; ++it) {
    const int idx = tid + it * 256;
    const int row = idx & 63;      // batch
    const int u   = idx >> 6;      // unit within tile
    const int j   = j0 + u;
    const float* sr = &stage[row * 65];
    float ip = sr[u]      + jb.bias[j];
    float fp = sr[16 + u] + jb.bias[H_ + j];
    float gp = sr[32 + u] + jb.bias[2 * H_ + j];
    float op = sr[48 + u] + jb.bias[3 * H_ + j];
    float i_ = sigm_fast(ip);
    float f_ = sigm_fast(fp);
    float o_ = sigm_fast(op);
    float g_ = tanh_fast(gp);
    const long ci = (long)row * H_ + j;
    float cn = f_ * jb.c[ci] + i_ * g_;
    jb.c[ci] = cn;
    float hn = o_ * tanh_fast(cn);
    jb.hout[ci] = f2bf(hn);
    if (jb.out32) jb.out32[(long)row * jb.outStride + j] = hn;
  }
}

// --------------------------- finalize --------------------------------------
__global__ void finalize_kernel(const unsigned short* __restrict__ h0T,
                                const unsigned short* __restrict__ h1T,
                                const float* __restrict__ cst,
                                float* __restrict__ outHF,
                                float* __restrict__ outCF) {
  int idx = blockIdx.x * blockDim.x + threadIdx.x;
  if (idx >= B_ * H_) return;
  union { unsigned int u; float f; } a, b;
  a.u = ((unsigned int)h0T[idx]) << 16;
  b.u = ((unsigned int)h1T[idx]) << 16;
  outHF[idx]            = a.f;
  outHF[B_ * H_ + idx]  = b.f;
  outCF[idx]            = cst[idx];
  outCF[B_ * H_ + idx]  = cst[B_ * H_ + idx];
}

// --------------------------- launcher --------------------------------------
extern "C" void kernel_launch(void* const* d_in, const int* in_sizes, int n_in,
                              void* d_out, int out_size, void* d_ws, size_t ws_size,
                              hipStream_t stream) {
  (void)in_sizes; (void)n_in; (void)out_size; (void)ws_size;

  const float* inputs = (const float*)d_in[0];
  const float* h0  = (const float*)d_in[1];
  const float* c0  = (const float*)d_in[2];
  const float* Wx0 = (const float*)d_in[3];
  const float* bx0 = (const float*)d_in[4];
  const float* Wh0 = (const float*)d_in[5];
  const float* bh0 = (const float*)d_in[6];
  const float* Wx1 = (const float*)d_in[7];
  const float* bx1 = (const float*)d_in[8];
  const float* Wh1 = (const float*)d_in[9];
  const float* bh1 = (const float*)d_in[10];

  // workspace layout (bytes); total ~135 MB
  char* ws = (char*)d_ws;
  const size_t WCAT_B = (size_t)FOURH * KTOT * 2;          // 16 MB
  const size_t BIAS_B = (size_t)FOURH * 4;                 // 16 KB
  const size_t XBF_B  = (size_t)B_ * T_ * IN_ * 2;         // 33.5 MB
  const size_t HSEQ_B = (size_t)(T_ + 1) * B_ * H_ * 2;    // 33.7 MB

  unsigned short* wcat0 = (unsigned short*)(ws);
  unsigned short* wcat1 = (unsigned short*)(ws + WCAT_B);
  float*          bias0 = (float*)(ws + 2 * WCAT_B);
  float*          bias1 = (float*)(ws + 2 * WCAT_B + BIAS_B);
  unsigned short* xbf   = (unsigned short*)(ws + 2 * WCAT_B + 2 * BIAS_B);
  unsigned short* h0seq = (unsigned short*)(ws + 2 * WCAT_B + 2 * BIAS_B + XBF_B);
  unsigned short* h1seq = (unsigned short*)(ws + 2 * WCAT_B + 2 * BIAS_B + XBF_B + HSEQ_B);
  float*          cst   = (float*)(ws + 2 * WCAT_B + 2 * BIAS_B + XBF_B + 2 * HSEQ_B);

  {
    long total = (long)FOURH * KTOT;
    int blocks = (int)((total + 255) / 256);
    prep_weights_kernel<<<blocks, 256, 0, stream>>>(Wx0, Wh0, bx0, bh0, wcat0, bias0);
    prep_weights_kernel<<<blocks, 256, 0, stream>>>(Wx1, Wh1, bx1, bh1, wcat1, bias1);
  }
  {
    long total = (long)B_ * T_ * IN_;
    prep_x_kernel<<<(int)((total + 255) / 256), 256, 0, stream>>>(inputs, xbf, total);
  }
  prep_state_kernel<<<(B_ * H_ + 255) / 256, 256, 0, stream>>>(h0, c0, h0seq, h1seq, cst);

  float* out = (float*)d_out;
  const long BH = (long)B_ * H_;

  // Software-pipelined: step s runs layer0(t=s) and layer1(t=s-1) in ONE launch.
  for (int s = 0; s <= T_; ++s) {
    StepJob jobs[2];
    int count = 0;
    if (s < T_) {  // layer 0 at time t=s
      jobs[count++] = StepJob{
          wcat0, bias0,
          xbf + (long)s * IN_, (long)T_ * IN_,
          h0seq + (long)s * BH, h0seq + (long)(s + 1) * BH,
          cst, nullptr, 0};
    }
    if (s >= 1) {  // layer 1 at time t=s-1
      const int t = s - 1;
      jobs[count++] = StepJob{
          wcat1, bias1,
          h0seq + (long)(t + 1) * BH, (long)H_,
          h1seq + (long)t * BH, h1seq + (long)(t + 1) * BH,
          cst + BH, out + (long)t * H_, (long)T_ * H_};
    }
    if (count == 1) jobs[1] = jobs[0];
    dim3 grid(H_ / 16, count);
    lstm_step_kernel<<<grid, 256, 0, stream>>>(jobs[0], jobs[1]);
  }

  finalize_kernel<<<(B_ * H_ + 255) / 256, 256, 0, stream>>>(
      h0seq + (long)T_ * BH, h1seq + (long)T_ * BH, cst,
      out + (long)B_ * T_ * H_,
      out + (long)B_ * T_ * H_ + 2 * BH);
}
